// DPPSampler_11845519802581
// MI455X (gfx1250) — matrix operand order repaired and checked by reading, not compiled, not run on hardware
//
#include <hip/hip_runtime.h>
#include <hip/hip_bf16.h>
#include <math.h>

typedef __attribute__((ext_vector_type(16))) _Float16 v16h;
typedef __attribute__((ext_vector_type(8)))  _Float16 v8h;
typedef __attribute__((ext_vector_type(4)))  _Float16 v4h;
typedef __attribute__((ext_vector_type(8)))  float    v8f;

#define LAMBDA_REG 0.01f
#define KMAX       256
#define NWG_GREEDY 32
#define TPB_GREEDY 256    // one owned column per thread
#define COLS       256    // columns per workgroup (NWG_GREEDY * COLS == N)
#define CSTR       260    // padded LDS row stride (520B: 8B-aligned, bank-spread)
#define NCTR       1024   // grid-barrier counters (2 per greedy iteration)

// ---------------------------------------------------------------------------
// Prep: g_i = f_i * sqrt(q_i)/||f_i||, split into f16 hi/lo; zero barrier ctrs
// ---------------------------------------------------------------------------
__global__ __launch_bounds__(256) void prep_kernel(
    const float* __restrict__ F, const float* __restrict__ q,
    _Float16* __restrict__ Ghi, _Float16* __restrict__ Glo,
    int* __restrict__ ctrs, int D) {
  const int row = blockIdx.x;
  const int tid = threadIdx.x;
  __shared__ float red[256];
  const float* frow = F + (size_t)row * D;
  float s = 0.f;
  for (int c = tid; c < D; c += 256) { float v = frow[c]; s += v * v; }
  red[tid] = s;
  __syncthreads();
  for (int o = 128; o > 0; o >>= 1) {
    if (tid < o) red[tid] += red[tid + o];
    __syncthreads();
  }
  const float scale = sqrtf(q[row]) / sqrtf(red[0]);
  _Float16* hrow = Ghi + (size_t)row * D;
  _Float16* lrow = Glo + (size_t)row * D;
  for (int c = tid; c < D; c += 256) {
    float g = frow[c] * scale;
    _Float16 hi = (_Float16)g;
    hrow[c] = hi;
    lrow[c] = (_Float16)(g - (float)hi);
  }
  if (row == 0) {
    for (int i = tid; i < NCTR; i += 256) ctrs[i] = 0;
  }
}

// ---------------------------------------------------------------------------
// GEMM: L = G*G^T + lambda*I via split-f16 WMMA (hi*hi + hi*lo + lo*hi).
// One wave computes one 64x64 macro-tile (4x4 grid of 16x16 WMMA tiles).
// Symmetry: only macro-tiles with Tm <= Tn computed; each tile stored in
// both orientations (mirror store is a contiguous b128 per lane).
// ---------------------------------------------------------------------------
__device__ inline v16h frag_a(const _Float16* p, int h) {
  // lane (m, h): halfs[0..7] = A[m][kb + h*8 .. +7], halfs[8..15] = A[m][kb+16+h*8 .. +7]
  v8h a = *(const v8h*)(p + h * 8);
  v8h b = *(const v8h*)(p + 16 + h * 8);
  return __builtin_shufflevector(a, b, 0,1,2,3,4,5,6,7,8,9,10,11,12,13,14,15);
}
__device__ inline v16h frag_b(const _Float16* p, int h) {
  // lane (n, h): halfs[0..7] = B[kb + h*8 .. +7][n], halfs[8..15] = B[kb+16+h*8 .. +7][n]
  v8h a = *(const v8h*)(p + h * 8);
  v8h b = *(const v8h*)(p + 16 + h * 8);
  return __builtin_shufflevector(a, b, 0,1,2,3,4,5,6,7,8,9,10,11,12,13,14,15);
}

__global__ __launch_bounds__(32) void gemm_kernel(
    const _Float16* __restrict__ Ghi, const _Float16* __restrict__ Glo,
    _Float16* __restrict__ L, int N, int D) {
  const int Tn = blockIdx.x;          // macro col (64 wide)
  const int Tm = blockIdx.y;          // macro row
  if (Tm > Tn) return;                // symmetric: upper triangle only
  const int lane = threadIdx.x & 31;
  const int m = lane & 15;
  const int h = lane >> 4;

  size_t offA[4], offB[4];
#pragma unroll
  for (int i = 0; i < 4; ++i) {
    offA[i] = (size_t)(Tm * 64 + i * 16 + m) * D;
    offB[i] = (size_t)(Tn * 64 + i * 16 + m) * D;
  }

  v8f acc[4][4] = {};
  for (int kb = 0; kb < D; kb += 32) {
    v16h ah[4], al[4], bh[4], bl[4];
#pragma unroll
    for (int i = 0; i < 4; ++i) {
      ah[i] = frag_a(Ghi + offA[i] + kb, h);
      al[i] = frag_a(Glo + offA[i] + kb, h);
      bh[i] = frag_b(Ghi + offB[i] + kb, h);
      bl[i] = frag_b(Glo + offB[i] + kb, h);
    }
#pragma unroll
    for (int i = 0; i < 4; ++i) {
#pragma unroll
      for (int j = 0; j < 4; ++j) {
        acc[i][j] = __builtin_amdgcn_wmma_f32_16x16x32_f16(false, ah[i], false, bh[j], (short)0, acc[i][j], false, false);
        acc[i][j] = __builtin_amdgcn_wmma_f32_16x16x32_f16(false, ah[i], false, bl[j], (short)0, acc[i][j], false, false);
        acc[i][j] = __builtin_amdgcn_wmma_f32_16x16x32_f16(false, al[i], false, bh[j], (short)0, acc[i][j], false, false);
      }
    }
  }

#pragma unroll
  for (int i = 0; i < 4; ++i) {
#pragma unroll
    for (int j = 0; j < 4; ++j) {
      const int colBase = Tn * 64 + j * 16 + m;       // this lane's column
      const int rowBase = Tm * 64 + i * 16 + 8 * h;   // 8 contiguous rows
      v8h pack;
#pragma unroll
      for (int r = 0; r < 8; ++r) {
        const int row = rowBase + r;
        float v = acc[i][j][r];
        if (row == colBase) v += LAMBDA_REG;
        L[(size_t)row * N + colBase] = (_Float16)v;   // normal orientation
        pack[r] = (_Float16)v;
      }
      // mirror (transpose) store: contiguous 8 halfs = one b128
      *(v8h*)(L + (size_t)colBase * N + rowBase) = pack;
    }
  }
}

// ---------------------------------------------------------------------------
// Persistent greedy DPP MAP: 32 WGs, cis resident in ~133KB LDS (f16),
// d_n in registers, atomic grid barrier (fresh counter per instance).
// ---------------------------------------------------------------------------
__device__ inline void grid_barrier(int* ctr, int count) {
  __syncthreads();
  if (threadIdx.x == 0) {
    __threadfence();
    __hip_atomic_fetch_add(ctr, 1, __ATOMIC_ACQ_REL, __HIP_MEMORY_SCOPE_AGENT);
    while (__hip_atomic_load(ctr, __ATOMIC_ACQUIRE, __HIP_MEMORY_SCOPE_AGENT) < count) {
      __builtin_amdgcn_s_sleep(1);
    }
    __threadfence();
  }
  __syncthreads();
}

__global__ __launch_bounds__(TPB_GREEDY, 1) void greedy_kernel(
    const _Float16* __restrict__ L, const float* __restrict__ q,
    const int* __restrict__ kptr, int* __restrict__ outIdx,
    float* __restrict__ pairV, int* __restrict__ pairI,
    float* __restrict__ cjG, int* __restrict__ ctrs, int N) {
  __shared__ _Float16 cis[COLS * CSTR];        // [local col][t], padded rows
  __shared__ float cj[KMAX];
  __shared__ float redV[TPB_GREEDY / 32];
  __shared__ int   redI[TPB_GREEDY / 32];
  __shared__ int   jSh;
  __shared__ float djSh;

  const int tid = threadIdx.x;
  const int wg  = blockIdx.x;
  const int n   = wg * COLS + tid;             // global column owned by thread
  float dloc = q[n] + LAMBDA_REG;              // L_nn = q_n + lambda
  int kk = *kptr;
  if (kk > KMAX) kk = KMAX;

  for (int t = 0; t < kk; ++t) {
    // ---- local argmax over owned columns (first-max tie-break) ----
    float v = dloc; int idx = n;
    for (int o = 16; o > 0; o >>= 1) {
      float ov = __shfl_down(v, o, 32);
      int   oi = __shfl_down(idx, o, 32);
      if (ov > v || (ov == v && oi < idx)) { v = ov; idx = oi; }
    }
    if ((tid & 31) == 0) { redV[tid >> 5] = v; redI[tid >> 5] = idx; }
    __syncthreads();
    if (tid == 0) {
      for (int w = 1; w < TPB_GREEDY / 32; ++w)
        if (redV[w] > v || (redV[w] == v && redI[w] < idx)) { v = redV[w]; idx = redI[w]; }
      pairV[wg] = v; pairI[wg] = idx;
    }
    grid_barrier(&ctrs[2 * t], NWG_GREEDY);

    // ---- global argmax, owner publishes column cis[:, j] ----
    if (tid == 0) {
      float bv = pairV[0]; int bi = pairI[0];
      for (int w = 1; w < NWG_GREEDY; ++w) {
        float wv = pairV[w]; int wi = pairI[w];
        if (wv > bv || (wv == bv && wi < bi)) { bv = wv; bi = wi; }
      }
      jSh = bi; djSh = bv;
    }
    __syncthreads();
    const int   j    = jSh;
    const float rsdj = rsqrtf(djSh);
    if ((j >> 8) == wg && tid < t)
      cjG[tid] = (float)cis[(size_t)(j & (COLS - 1)) * CSTR + tid];
    if (wg == 0 && tid == 0) outIdx[t] = j;
    grid_barrier(&ctrs[2 * t + 1], NWG_GREEDY);

    // ---- e = (L[j,:] - cis[:,j] @ cis) * rsqrt(d_j); update d, append row ----
    if (tid < t) cj[tid] = cjG[tid];
    __syncthreads();
    float acc = 0.f;
    _Float16* crow = cis + (size_t)tid * CSTR;
    int tt = 0;
    for (; tt + 4 <= t; tt += 4) {               // v4h-vectorized inner product
      v4h c4 = *(const v4h*)(crow + tt);
      acc += cj[tt]     * (float)c4[0];
      acc += cj[tt + 1] * (float)c4[1];
      acc += cj[tt + 2] * (float)c4[2];
      acc += cj[tt + 3] * (float)c4[3];
    }
    for (; tt < t; ++tt) acc += cj[tt] * (float)crow[tt];
    const float Lval = (float)L[(size_t)j * N + n];   // lambda already on diag
    const float e = (Lval - acc) * rsdj;
    crow[t] = (_Float16)e;
    dloc -= e * e;
    if (n == j) dloc = -__builtin_inff();             // mask selected item
  }
}

// ---------------------------------------------------------------------------
extern "C" void kernel_launch(void* const* d_in, const int* in_sizes, int n_in,
                              void* d_out, int out_size, void* d_ws, size_t ws_size,
                              hipStream_t stream) {
  const float* F    = (const float*)d_in[0];
  const float* q    = (const float*)d_in[1];
  const int*   kptr = (const int*)d_in[2];
  const int N = in_sizes[1];            // 8192
  const int D = in_sizes[0] / N;        // 512

  char* w = (char*)d_ws;
  const size_t szL = (size_t)N * N * sizeof(_Float16);   // 128 MB
  const size_t szG = (size_t)N * D * sizeof(_Float16);   // 8 MB each
  _Float16* Lf  = (_Float16*)w;
  _Float16* Ghi = (_Float16*)(w + szL);
  _Float16* Glo = (_Float16*)(w + szL + szG);
  char* p = w + szL + 2 * szG;
  float* pairV = (float*)p;            p += 64 * sizeof(float);
  int*   pairI = (int*)p;              p += 64 * sizeof(int);
  float* cjG   = (float*)p;            p += KMAX * sizeof(float);
  int*   ctrs  = (int*)p;
  int*   out   = (int*)d_out;

  prep_kernel<<<N, 256, 0, stream>>>(F, q, Ghi, Glo, ctrs, D);

  dim3 ggrid(N / 64, N / 64);           // 128 x 128 macro-tiles, upper triangle active
  gemm_kernel<<<ggrid, 32, 0, stream>>>(Ghi, Glo, Lf, N, D);

  greedy_kernel<<<NWG_GREEDY, TPB_GREEDY, 0, stream>>>(Lf, q, kptr, out,
                                                       pairV, pairI, cjG, ctrs, N);
}
